// TransformerBlock_87119116632100
// MI455X (gfx1250) — matrix operand order repaired and checked by reading, not compiled, not run on hardware
//
#include <hip/hip_runtime.h>
#include <cstdint>

typedef __bf16 bf16;
typedef __attribute__((ext_vector_type(16))) __bf16 v16bf;
typedef __attribute__((ext_vector_type(8)))  __bf16 v8bf;
typedef __attribute__((ext_vector_type(8)))  float  v8f;

#define DD   1024
#define TT   512
#define BB   4
#define NTOK 2048
#define HQ_  16
#define HKV_ 4
#define KD   64
#define VD   64
#define EE   8
#define HID_ 2048
#define CAPN 256

// ---------- WMMA helpers (CDNA5 16x16x32 bf16, f32 accum) ----------
__device__ __forceinline__ v16bf cat8(v8bf a, v8bf b){
  v16bf r;
#pragma unroll
  for (int i = 0; i < 8; ++i){ r[i] = a[i]; r[i + 8] = b[i]; }
  return r;
}
__device__ __forceinline__ v8bf ld8(const bf16* p){ return *reinterpret_cast<const v8bf*>(p); }
__device__ __forceinline__ v8f wmma_bf16(v16bf a, v16bf b, v8f c){
  return __builtin_amdgcn_wmma_f32_16x16x32_bf16(false, a, false, b, (short)0, c, false, false);
}

// ---------- async global->LDS copy (gfx1250 GLOBAL_LOAD_ASYNC_TO_LDS_B128) ----------
#if defined(__has_builtin)
#  if __has_builtin(__builtin_amdgcn_global_load_async_to_lds_b128)
#    define ASYNC_CP 1
#  endif
#endif

typedef __attribute__((ext_vector_type(4))) int i32x4;
typedef __attribute__((address_space(1))) i32x4 gl_i32x4;  // global AS pointee
typedef __attribute__((address_space(3))) i32x4 ds_i32x4;  // LDS AS pointee

__device__ __forceinline__ void cp16(const bf16* g, bf16* l){
#ifdef ASYNC_CP
  __builtin_amdgcn_global_load_async_to_lds_b128((gl_i32x4*)g, (ds_i32x4*)l, 0, 0);
#else
  *reinterpret_cast<v8bf*>(l) = *reinterpret_cast<const v8bf*>(g);
#endif
}
__device__ __forceinline__ void cp_wait(){
#ifdef ASYNC_CP
#  if __has_builtin(__builtin_amdgcn_s_wait_asynccnt)
  __builtin_amdgcn_s_wait_asynccnt(0);
#  else
  asm volatile("s_wait_asynccnt 0" ::: "memory");
#  endif
#endif
}

// ---------- RMSNorm: fp32 in -> bf16 (and optional fp32) out ----------
__global__ void rmsnorm_kernel(const float* __restrict__ x, const float* __restrict__ g,
                               bf16* __restrict__ outb, float* __restrict__ outf){
  int n = blockIdx.x;
  const float* row = x + (size_t)n * DD;
  float s = 0.f;
  for (int d = threadIdx.x; d < DD; d += 256){ float v = row[d]; s += v * v; }
#pragma unroll
  for (int o = 16; o > 0; o >>= 1) s += __shfl_xor(s, o, 32);
  __shared__ float red[8];
  int wv = threadIdx.x >> 5;
  if ((threadIdx.x & 31) == 0) red[wv] = s;
  __syncthreads();
  if (wv == 0){
    float t = (threadIdx.x < 8) ? red[threadIdx.x] : 0.f;
#pragma unroll
    for (int o = 4; o > 0; o >>= 1) t += __shfl_xor(t, o, 32);
    if (threadIdx.x == 0) red[0] = t;
  }
  __syncthreads();
  float rs = rsqrtf(red[0] * (1.0f / DD) + 1e-6f);
  for (int d = threadIdx.x; d < DD; d += 256){
    float v = row[d] * rs * g[d];
    outb[(size_t)n * DD + d] = (bf16)v;
    if (outf) outf[(size_t)n * DD + d] = v;
  }
}

// ---------- Generic GEMM: C(MxN) = A_bf16(MxK) * Bw_f32->bf16(KxN) [+R] ----------
// Block tile 128x64, 8 waves: wave = 16-row strip x 64 cols (4 accumulators).
template<bool ADD_RESID>
__global__ void gemm_bf16w_kernel(const bf16* __restrict__ A, const float* __restrict__ Bw,
                                  float* __restrict__ C, const float* __restrict__ R,
                                  int M, int Ncol, int Kd){
  __shared__ bf16 As[128][40];  // 80B pitch -> 16B-aligned subrows
  __shared__ bf16 Bs[64][40];   // transposed weight tile: Bs[col][k]
  const int tid = threadIdx.x;
  const int lane = tid & 31, wave = tid >> 5;
  const int l15 = lane & 15, hi = lane >> 4;
  const int n0 = blockIdx.x * 64, m0 = blockIdx.y * 128;
  v8f acc[4] = {{}, {}, {}, {}};
  for (int k0 = 0; k0 < Kd; k0 += 32){
    // stage A 128x32 bf16 (async to LDS when available)
#pragma unroll
    for (int j = 0; j < 2; ++j){
      int c = tid * 2 + j;                 // 512 chunks of 8 elems
      int r = c >> 2, kg = (c & 3) * 8;
      cp16(A + (size_t)(m0 + r) * Kd + k0 + kg, &As[r][kg]);
    }
    { // stage B 32x64 transposed with f32->bf16 convert
      int krow = tid >> 3, cg = (tid & 7) * 8;
      const float* src = Bw + (size_t)(k0 + krow) * Ncol + n0 + cg;
#pragma unroll
      for (int j = 0; j < 8; ++j) Bs[cg + j][krow] = (bf16)src[j];
      if (k0 + 32 < Kd)
        __builtin_prefetch(Bw + (size_t)(k0 + 32 + krow) * Ncol + n0 + cg, 0, 0);
    }
    cp_wait();
    __syncthreads();
    const bf16* ap = &As[wave * 16 + l15][hi * 8];
    v16bf af = cat8(ld8(ap), ld8(ap + 16));
#pragma unroll
    for (int ct = 0; ct < 4; ++ct){
      const bf16* bp = &Bs[ct * 16 + l15][hi * 8];
      acc[ct] = wmma_bf16(af, cat8(ld8(bp), ld8(bp + 16)), acc[ct]);
    }
    __syncthreads();
  }
#pragma unroll
  for (int ct = 0; ct < 4; ++ct){
#pragma unroll
    for (int r = 0; r < 8; ++r){
      int row = m0 + wave * 16 + r + hi * 8;
      int col = n0 + ct * 16 + l15;
      float vv = acc[ct][r];
      if (ADD_RESID) vv += R[(size_t)row * Ncol + col];
      C[(size_t)row * Ncol + col] = vv;
    }
  }
}

// ---------- Rotary embedding: fp32 in -> bf16 out ----------
__global__ void rotary_kernel(const float* __restrict__ in, bf16* __restrict__ out,
                              int H, int total){
  int i = blockIdx.x * blockDim.x + threadIdx.x;
  if (i >= total) return;
  int kp = i & 31;
  int nh = i >> 5;
  int n = nh / H;
  int t = n & (TT - 1);
  size_t base = (size_t)nh * 64;
  float ang = (float)t * __expf(-(float)kp * 0.2878231366f); // 10000^(-kp/32)
  float sn, cs; __sincosf(ang, &sn, &cs);
  float x1 = in[base + kp], x2 = in[base + kp + 32];
  out[base + kp]      = (bf16)(x1 * cs - x2 * sn);
  out[base + kp + 32] = (bf16)(x2 * cs + x1 * sn);
}

__global__ void f2bf_kernel(const float* __restrict__ in, bf16* __restrict__ out, int total){
  int i = blockIdx.x * blockDim.x + threadIdx.x;
  if (i < total) out[i] = (bf16)in[i];
}

// ---------- Flash attention: grid(T/128, HQ, B), 8 waves x 16 q-rows ----------
__global__ void attn_kernel(const bf16* __restrict__ q, const bf16* __restrict__ k,
                            const bf16* __restrict__ v, bf16* __restrict__ o){
  __shared__ bf16 Vt[64][40];      // V tile transposed: Vt[vcol][krow]
  __shared__ bf16 Pw[8][16][32];   // per-wave P relayout scratch
  const int tid = threadIdx.x, lane = tid & 31, wave = tid >> 5;
  const int l15 = lane & 15, hi = lane >> 4, hi8 = hi * 8;
  const int qt = blockIdx.x, hq = blockIdx.y, bb = blockIdx.z;
  const int kv = hq & (HKV_ - 1);
  const int qbase = qt * 128 + wave * 16;
  const bf16* qrow = q + ((size_t)(bb * TT + qbase + l15) * HQ_ + hq) * KD;
  v16bf qa0 = cat8(ld8(qrow + hi8),      ld8(qrow + hi8 + 16));
  v16bf qa1 = cat8(ld8(qrow + 32 + hi8), ld8(qrow + 32 + hi8 + 16));
  v8f o0 = {}, o1 = {}, o2 = {}, o3 = {};
  float rm[8], rl[8];
#pragma unroll
  for (int r = 0; r < 8; ++r){ rm[r] = -3.0e38f; rl[r] = 0.f; }
  const int jend = qt * 128 + 128;  // uniform trip count over whole block
  for (int j32 = 0; j32 < jend; j32 += 32){
    __syncthreads();
    { // stage V^T 32x64
      int krow = tid >> 3, cg = (tid & 7) * 8;
      v8bf vv = ld8(v + ((size_t)(bb * TT + j32 + krow) * HKV_ + kv) * VD + cg);
#pragma unroll
      for (int j = 0; j < 8; ++j) Vt[cg + j][krow] = vv[j];
    }
    __syncthreads();
    v8f s0 = {}, s1 = {};
#pragma unroll
    for (int x = 0; x < 2; ++x){
      const bf16* kr = k + ((size_t)(bb * TT + j32 + x * 16 + l15) * HKV_ + kv) * KD;
      v16bf b0 = cat8(ld8(kr + hi * 8),      ld8(kr + hi * 8 + 16));
      v16bf b1 = cat8(ld8(kr + 32 + hi * 8), ld8(kr + 32 + hi * 8 + 16));
      if (x == 0){ s0 = wmma_bf16(qa0, b0, s0); s0 = wmma_bf16(qa1, b1, s0); }
      else       { s1 = wmma_bf16(qa0, b0, s1); s1 = wmma_bf16(qa1, b1, s1); }
    }
#pragma unroll
    for (int r = 0; r < 8; ++r){
      int row = qbase + r + hi8;
      int c0 = j32 + l15, c1 = c0 + 16;
      float a  = (c0 <= row) ? s0[r] * 0.125f : -3.0e38f;
      float b2 = (c1 <= row) ? s1[r] * 0.125f : -3.0e38f;
      float mloc = fmaxf(a, b2);
#pragma unroll
      for (int off = 1; off < 16; off <<= 1) mloc = fmaxf(mloc, __shfl_xor(mloc, off, 32));
      float mnew = fmaxf(rm[r], mloc);
      float alpha = __expf(rm[r] - mnew);
      float p0 = __expf(a - mnew), p1 = __expf(b2 - mnew);
      float ls = p0 + p1;
#pragma unroll
      for (int off = 1; off < 16; off <<= 1) ls += __shfl_xor(ls, off, 32);
      rl[r] = rl[r] * alpha + ls;
      rm[r] = mnew;
      o0[r] *= alpha; o1[r] *= alpha; o2[r] *= alpha; o3[r] *= alpha;
      Pw[wave][r + hi8][l15]      = (bf16)p0;
      Pw[wave][r + hi8][l15 + 16] = (bf16)p1;
    }
    asm volatile("s_wait_dscnt 0" ::: "memory");  // intra-wave LDS RAW fence
    const bf16* pp = &Pw[wave][l15][hi8];
    v16bf pf = cat8(ld8(pp), ld8(pp + 16));
    const bf16* vb0 = &Vt[ 0 + l15][hi * 8];
    const bf16* vb1 = &Vt[16 + l15][hi * 8];
    const bf16* vb2 = &Vt[32 + l15][hi * 8];
    const bf16* vb3 = &Vt[48 + l15][hi * 8];
    o0 = wmma_bf16(pf, cat8(ld8(vb0), ld8(vb0 + 16)), o0);
    o1 = wmma_bf16(pf, cat8(ld8(vb1), ld8(vb1 + 16)), o1);
    o2 = wmma_bf16(pf, cat8(ld8(vb2), ld8(vb2 + 16)), o2);
    o3 = wmma_bf16(pf, cat8(ld8(vb3), ld8(vb3 + 16)), o3);
  }
#pragma unroll
  for (int r = 0; r < 8; ++r){
    int row = qbase + r + hi8;
    float inv = 1.0f / fmaxf(rl[r], 1e-20f);
    size_t ob = (size_t)(bb * TT + row) * (HQ_ * VD) + hq * VD;
    o[ob +  0 + l15] = (bf16)(o0[r] * inv);
    o[ob + 16 + l15] = (bf16)(o1[r] * inv);
    o[ob + 32 + l15] = (bf16)(o2[r] * inv);
    o[ob + 48 + l15] = (bf16)(o3[r] * inv);
  }
}

// ---------- Router: one wave per token, 8-expert logits + top-2 softmax ----------
__global__ void router_kernel(const float* __restrict__ xn, const float* __restrict__ rw,
                              const float* __restrict__ rb, int* __restrict__ i0,
                              int* __restrict__ i1, float* __restrict__ p0,
                              float* __restrict__ p1){
  int n = blockIdx.x * 8 + (threadIdx.x >> 5);
  int lane = threadIdx.x & 31;
  float acc[EE];
#pragma unroll
  for (int e = 0; e < EE; ++e) acc[e] = 0.f;
  const float* row = xn + (size_t)n * DD;
  for (int d = lane; d < DD; d += 32){
    float xv = row[d];
#pragma unroll
    for (int e = 0; e < EE; ++e) acc[e] += xv * rw[d * EE + e];
  }
#pragma unroll
  for (int e = 0; e < EE; ++e){
#pragma unroll
    for (int off = 16; off > 0; off >>= 1) acc[e] += __shfl_xor(acc[e], off, 32);
  }
  if (lane == 0){
    float b0 = -3e38f, b1 = -3e38f; int e0 = 0, e1 = 0;
#pragma unroll
    for (int e = 0; e < EE; ++e){
      float lg = acc[e] + rb[e];
      if (lg > b0){ b1 = b0; e1 = e0; b0 = lg; e0 = e; }
      else if (lg > b1){ b1 = lg; e1 = e; }
    }
    float s0 = 1.0f / (1.0f + __expf(b1 - b0));
    i0[n] = e0; i1[n] = e1; p0[n] = s0; p1[n] = 1.0f - s0;
  }
}

// ---------- Capacity routing (serial cumsum(cumsum) drop logic) ----------
__global__ void capacity_kernel(const int* __restrict__ i0, const int* __restrict__ i1,
                                const float* __restrict__ p0, const float* __restrict__ p1,
                                int* __restrict__ list, int* __restrict__ cnt,
                                float* __restrict__ wy, float* __restrict__ wx){
  if (threadIdx.x != 0 || blockIdx.x != 0) return;
  int c0[EE], c1[EE], cl[EE];
  for (int e = 0; e < EE; ++e){ c0[e] = 0; c1[e] = 0; cl[e] = 0; }
  for (int n = 0; n < NTOK; ++n){
    int e0 = i0[n], e1 = i1[n];
    int pos0 = ++c0[e0];
    int keep0 = pos0 < CAPN;
    int pos1 = c0[e1] + (++c1[e1]);
    int keep1 = pos1 < CAPN;
    int a0 = keep0 ? e0 : 0;
    int a1 = keep1 ? e1 : 0;
    int m = a0 > a1 ? a0 : a1;
    float s0 = p0[n], s1 = p1[n];
    float wyv = (keep0 ? s0 : 0.f) + (keep1 ? s1 : 0.f);
    wy[n] = wyv; wx[n] = s0 + s1 - wyv;
    list[m * NTOK + cl[m]++] = n;
  }
  for (int e = 0; e < EE; ++e) cnt[e] = cl[e];
}

// ---------- MoE FFN1: h = gelu((x@w2)*(x@w1)), 64-token gathered tiles ----------
// Block tile 64 tokens x 64 hidden cols; waves in 4x2 grid, 2+2 accumulators.
__global__ void moe_ffn1_kernel(const bf16* __restrict__ X, const float* __restrict__ w1,
                                const float* __restrict__ w2, const int* __restrict__ list,
                                const int* __restrict__ cnt, bf16* __restrict__ hbuf){
  const int e = blockIdx.z;
  const int ce = cnt[e];
  if ((int)blockIdx.x * 64 >= ce) return;
  __shared__ bf16 Xs[64][40];
  __shared__ bf16 B1s[64][40];
  __shared__ bf16 B2s[64][40];
  __shared__ int tokLds[64];
  const int tid = threadIdx.x, lane = tid & 31, wave = tid >> 5;
  const int l15 = lane & 15, hi = lane >> 4;
  const int n0 = blockIdx.y * 64;
  const int wr = wave & 3, wc = wave >> 2;
  if (tid < 64){
    int s = blockIdx.x * 64 + tid;
    tokLds[tid] = list[e * NTOK + (s < ce ? s : ce - 1)];
  }
  __syncthreads();
  v8f a1[2] = {{}, {}}, a2[2] = {{}, {}};
  for (int k0 = 0; k0 < DD; k0 += 32){
    { // stage gathered X 64x32
      int r = tid >> 2, kg = (tid & 3) * 8;
      cp16(X + (size_t)tokLds[r] * DD + k0 + kg, &Xs[r][kg]);
    }
    { // stage w1/w2 32x64 transposed with convert
      int krow = tid >> 3, cg = (tid & 7) * 8;
      const float* s1 = w1 + ((size_t)e * DD + k0 + krow) * HID_ + n0 + cg;
      const float* s2 = w2 + ((size_t)e * DD + k0 + krow) * HID_ + n0 + cg;
#pragma unroll
      for (int j = 0; j < 8; ++j){ B1s[cg + j][krow] = (bf16)s1[j]; B2s[cg + j][krow] = (bf16)s2[j]; }
      if (k0 + 32 < DD)
        __builtin_prefetch(w1 + ((size_t)e * DD + k0 + 32 + krow) * HID_ + n0 + cg, 0, 0);
    }
    cp_wait();
    __syncthreads();
    const bf16* ap = &Xs[wr * 16 + l15][hi * 8];
    v16bf af = cat8(ld8(ap), ld8(ap + 16));
#pragma unroll
    for (int ct = 0; ct < 2; ++ct){
      const bf16* b1p = &B1s[wc * 32 + ct * 16 + l15][hi * 8];
      const bf16* b2p = &B2s[wc * 32 + ct * 16 + l15][hi * 8];
      a1[ct] = wmma_bf16(af, cat8(ld8(b1p), ld8(b1p + 16)), a1[ct]);
      a2[ct] = wmma_bf16(af, cat8(ld8(b2p), ld8(b2p + 16)), a2[ct]);
    }
    __syncthreads();
  }
#pragma unroll
  for (int ct = 0; ct < 2; ++ct){
#pragma unroll
    for (int r = 0; r < 8; ++r){
      int srow = wr * 16 + r + hi * 8;
      int slot = blockIdx.x * 64 + srow;
      if (slot < ce){
        int tok = tokLds[srow];
        float u = a1[ct][r] * a2[ct][r];
        float g = 0.5f * u * (1.0f + erff(u * 0.70710678f));
        hbuf[(size_t)tok * HID_ + n0 + wc * 32 + ct * 16 + l15] = (bf16)g;
      }
    }
  }
}

// ---------- MoE FFN2: y = h@w3, fused final combine into d_out ----------
__global__ void moe_ffn2_kernel(const bf16* __restrict__ hbuf, const float* __restrict__ w3,
                                const int* __restrict__ list, const int* __restrict__ cnt,
                                const float* __restrict__ h32, const float* __restrict__ hn32,
                                const float* __restrict__ wy, const float* __restrict__ wx,
                                float* __restrict__ out){
  const int e = blockIdx.z;
  const int ce = cnt[e];
  if ((int)blockIdx.x * 64 >= ce) return;
  __shared__ bf16 As[64][40];
  __shared__ bf16 Bs[64][40];
  __shared__ int tokLds[64];
  const int tid = threadIdx.x, lane = tid & 31, wave = tid >> 5;
  const int l15 = lane & 15, hi = lane >> 4;
  const int n0 = blockIdx.y * 64;
  const int wr = wave & 3, wc = wave >> 2;
  if (tid < 64){
    int s = blockIdx.x * 64 + tid;
    tokLds[tid] = list[e * NTOK + (s < ce ? s : ce - 1)];
  }
  __syncthreads();
  v8f acc[2] = {{}, {}};
  for (int k0 = 0; k0 < HID_; k0 += 32){
    { // stage gathered h 64x32
      int r = tid >> 2, kg = (tid & 3) * 8;
      cp16(hbuf + (size_t)tokLds[r] * HID_ + k0 + kg, &As[r][kg]);
    }
    { // stage w3 32x64 transposed with convert
      int krow = tid >> 3, cg = (tid & 7) * 8;
      const float* s3 = w3 + ((size_t)e * HID_ + k0 + krow) * DD + n0 + cg;
#pragma unroll
      for (int j = 0; j < 8; ++j) Bs[cg + j][krow] = (bf16)s3[j];
      if (k0 + 32 < HID_)
        __builtin_prefetch(w3 + ((size_t)e * HID_ + k0 + 32 + krow) * DD + n0 + cg, 0, 0);
    }
    cp_wait();
    __syncthreads();
    const bf16* ap = &As[wr * 16 + l15][hi * 8];
    v16bf af = cat8(ld8(ap), ld8(ap + 16));
#pragma unroll
    for (int ct = 0; ct < 2; ++ct){
      const bf16* bp = &Bs[wc * 32 + ct * 16 + l15][hi * 8];
      acc[ct] = wmma_bf16(af, cat8(ld8(bp), ld8(bp + 16)), acc[ct]);
    }
    __syncthreads();
  }
#pragma unroll
  for (int ct = 0; ct < 2; ++ct){
#pragma unroll
    for (int r = 0; r < 8; ++r){
      int srow = wr * 16 + r + hi * 8;
      int slot = blockIdx.x * 64 + srow;
      if (slot < ce){
        int tok = tokLds[srow];
        int col = n0 + wc * 32 + ct * 16 + l15;
        size_t idx = (size_t)tok * DD + col;
        out[idx] = h32[idx] + wy[tok] * acc[ct][r] + wx[tok] * hn32[idx];
      }
    }
  }
}

// ---------- host side ----------
extern "C" void kernel_launch(void* const* d_in, const int* in_sizes, int n_in,
                              void* d_out, int out_size, void* d_ws, size_t ws_size,
                              hipStream_t stream){
  const float* x  = (const float*)d_in[0];
  const float* g1 = (const float*)d_in[1];
  const float* g2 = (const float*)d_in[2];
  const float* wq = (const float*)d_in[3];
  const float* wk = (const float*)d_in[4];
  const float* wv = (const float*)d_in[5];
  const float* wo = (const float*)d_in[6];
  const float* rw = (const float*)d_in[7];
  const float* rb = (const float*)d_in[8];
  const float* w1 = (const float*)d_in[9];
  const float* w2 = (const float*)d_in[10];
  const float* w3 = (const float*)d_in[11];
  float* out = (float*)d_out;

  char* p = (char*)d_ws;
  auto take = [&](size_t bytes) -> char* {
    char* r = p; p += (bytes + 255) & ~size_t(255); return r;
  };
  bf16*  xn_bf = (bf16*) take((size_t)NTOK * DD * 2);
  float* q32   = (float*)take((size_t)NTOK * 1024 * 4);
  float* k32   = (float*)take((size_t)NTOK * 256 * 4);
  float* v32   = (float*)take((size_t)NTOK * 256 * 4);
  bf16*  qb    = (bf16*) take((size_t)NTOK * 1024 * 2);
  bf16*  kb    = (bf16*) take((size_t)NTOK * 256 * 2);
  bf16*  vb    = (bf16*) take((size_t)NTOK * 256 * 2);
  bf16*  attnb = (bf16*) take((size_t)NTOK * 1024 * 2);
  float* h32   = (float*)take((size_t)NTOK * DD * 4);
  bf16*  hnb   = (bf16*) take((size_t)NTOK * DD * 2);
  float* hn32  = (float*)take((size_t)NTOK * DD * 4);
  int*   i0    = (int*)  take((size_t)NTOK * 4);
  int*   i1    = (int*)  take((size_t)NTOK * 4);
  float* p0    = (float*)take((size_t)NTOK * 4);
  float* p1    = (float*)take((size_t)NTOK * 4);
  float* wyb   = (float*)take((size_t)NTOK * 4);
  float* wxb   = (float*)take((size_t)NTOK * 4);
  int*   list  = (int*)  take((size_t)EE * NTOK * 4);
  int*   cnt   = (int*)  take(256);
  bf16*  hbuf  = (bf16*) take((size_t)NTOK * HID_ * 2);

  rmsnorm_kernel<<<NTOK, 256, 0, stream>>>(x, g1, xn_bf, nullptr);
  gemm_bf16w_kernel<false><<<dim3(1024/64, NTOK/128), 256, 0, stream>>>(xn_bf, wq, q32, nullptr, NTOK, 1024, DD);
  gemm_bf16w_kernel<false><<<dim3(256/64,  NTOK/128), 256, 0, stream>>>(xn_bf, wk, k32, nullptr, NTOK, 256,  DD);
  gemm_bf16w_kernel<false><<<dim3(256/64,  NTOK/128), 256, 0, stream>>>(xn_bf, wv, v32, nullptr, NTOK, 256,  DD);
  rotary_kernel<<<(NTOK * HQ_  * 32) / 256, 256, 0, stream>>>(q32, qb, HQ_,  NTOK * HQ_  * 32);
  rotary_kernel<<<(NTOK * HKV_ * 32) / 256, 256, 0, stream>>>(k32, kb, HKV_, NTOK * HKV_ * 32);
  f2bf_kernel<<<(NTOK * 256) / 256, 256, 0, stream>>>(v32, vb, NTOK * 256);
  attn_kernel<<<dim3(TT/128, HQ_, BB), 256, 0, stream>>>(qb, kb, vb, attnb);
  gemm_bf16w_kernel<true><<<dim3(1024/64, NTOK/128), 256, 0, stream>>>(attnb, wo, h32, x, NTOK, 1024, 1024);
  rmsnorm_kernel<<<NTOK, 256, 0, stream>>>(h32, g2, hnb, hn32);
  router_kernel<<<NTOK/8, 256, 0, stream>>>(hn32, rw, rb, i0, i1, p0, p1);
  capacity_kernel<<<1, 1, 0, stream>>>(i0, i1, p0, p1, list, cnt, wyb, wxb);
  moe_ffn1_kernel<<<dim3(NTOK/64, HID_/64, EE), 256, 0, stream>>>(hnb, w1, w2, list, cnt, hbuf);
  moe_ffn2_kernel<<<dim3(NTOK/64, DD/64, EE), 256, 0, stream>>>(hbuf, w3, list, cnt, h32, hn32, wyb, wxb, out);
  (void)in_sizes; (void)n_in; (void)out_size; (void)ws_size;
}
